// Attention_61409442398286
// MI455X (gfx1250) — matrix operand order repaired and checked by reading, not compiled, not run on hardware
//
#include <hip/hip_runtime.h>

// ---------------------------------------------------------------------------
// Decomposable attention for MI455X (gfx1250, wave32, WMMA).
// All heavy math runs through v_wmma_f32_16x16x32_bf16 (fp32 accumulate).
// ---------------------------------------------------------------------------

typedef __attribute__((ext_vector_type(16))) __bf16      v16bf;
typedef __attribute__((ext_vector_type(8)))  float       v8f;
typedef __attribute__((ext_vector_type(4)))  unsigned int u32x4;

#define DEV __device__ __forceinline__

DEV unsigned short f2bf(float x) {
  unsigned int u = __float_as_uint(x);
  u += 0x7FFFu + ((u >> 16) & 1u);           // round-to-nearest-even
  return (unsigned short)(u >> 16);
}

// A fragment (16x32 bf16, M x K): lane holds row m = lane&15.
// lanes 0-15: K chunks {0..7},{16..23}; lanes 16-31: {8..15},{24..31}.
// p points at &A[row][k0]; kh = (lane>>4)*8 (elements).
DEV v16bf load_a_frag(const unsigned short* p, int kh) {
  union { u32x4 u[2]; v16bf v; } f;
  f.u[0] = *(const u32x4*)(p + kh);
  f.u[1] = *(const u32x4*)(p + kh + 16);
  return f.v;
}

// B fragment (32x16 bf16, K x N) loaded from Bw[N][K] row-major (NT form):
// lane holds column n = lane&15; lanes 0-15: K chunks {0..7},{16..23}; lanes 16-31: {8..15},{24..31}
// (same as A). p points at &Bw[n][k0]; kq = (lane>>4)*8.
DEV v16bf load_b_frag(const unsigned short* p, int kq) {
  union { u32x4 u[2]; v16bf v; } f;
  f.u[0] = *(const u32x4*)(p + kq);
  f.u[1] = *(const u32x4*)(p + kq + 16);
  return f.v;
}

// ---------------------------------------------------------------------------
// C[M,N](batch) = act( A[M,K] * Bw[N,K]^T + bias[N] )
// Block: 128 threads (4 waves, 2x2), tile 128x128. Wave tile 64x64 (4x4 WMMA).
// M,N multiples of 128; K multiple of 32. Strides s* in elements.
// ---------------------------------------------------------------------------
template <int RELU, int BF16OUT, int BIAS>
__global__ __launch_bounds__(128) void gemm_nt(
    const unsigned short* __restrict__ A,  size_t sA,
    const unsigned short* __restrict__ Bw, size_t sB,
    void* __restrict__ Cv, size_t sC,
    const float* __restrict__ bias,
    int lda, int ldb, int ldc, int K)
{
  const int lane = threadIdx.x & 31;
  const int w    = threadIdx.x >> 5;
  const int wm   = w >> 1, wn = w & 1;
  const int bz   = blockIdx.z;

  const unsigned short* Ab = A  + (size_t)bz * sA;
  const unsigned short* Bb = Bw + (size_t)bz * sB;

  const int m_wave = blockIdx.y * 128 + wm * 64;
  const int n_wave = blockIdx.x * 128 + wn * 64;

  const int kh = (lane >> 4) * 8;    // A chunk select (elements)
  const int kq = (lane >> 4) * 8;    // B chunk select (elements)
  const int rl = lane & 15;

  const unsigned short* aRow[4];
  const unsigned short* bRow[4];
#pragma unroll
  for (int i = 0; i < 4; ++i) aRow[i] = Ab + (size_t)(m_wave + 16 * i + rl) * lda;
#pragma unroll
  for (int j = 0; j < 4; ++j) bRow[j] = Bb + (size_t)(n_wave + 16 * j + rl) * ldb;

  const v8f vzero = {};
  v8f acc[4][4];
#pragma unroll
  for (int i = 0; i < 4; ++i)
#pragma unroll
    for (int j = 0; j < 4; ++j) acc[i][j] = vzero;

  for (int k0 = 0; k0 < K; k0 += 32) {
    // prefetch K tile 4 iterations ahead (global_prefetch_b8 path)
    __builtin_prefetch(aRow[0] + k0 + 128, 0, 1);
    __builtin_prefetch(bRow[0] + k0 + 128, 0, 1);

    v16bf af[4];
#pragma unroll
    for (int i = 0; i < 4; ++i) af[i] = load_a_frag(aRow[i] + k0, kh);
#pragma unroll
    for (int j = 0; j < 4; ++j) {
      v16bf bfr = load_b_frag(bRow[j] + k0, kq);
#pragma unroll
      for (int i = 0; i < 4; ++i) {
        acc[i][j] = __builtin_amdgcn_wmma_f32_16x16x32_bf16(
            false, af[i], false, bfr, (short)0, acc[i][j], false, false);
      }
    }
  }

  // Epilogue. C element (m,n): VGPR v -> m = v + 8*(lane>=16), n = lane&15.
  const int mh = 8 * (lane >> 4);
  const int nl = lane & 15;
#pragma unroll
  for (int j = 0; j < 4; ++j) {
    const int n  = n_wave + 16 * j + nl;
    const float bv = BIAS ? bias[n] : 0.0f;
#pragma unroll
    for (int i = 0; i < 4; ++i) {
      const int mbase = m_wave + 16 * i + mh;
#pragma unroll
      for (int v = 0; v < 8; ++v) {
        float val = acc[i][j][v] + bv;
        if (RELU) val = fmaxf(val, 0.0f);
        const size_t idx = (size_t)bz * sC + (size_t)(mbase + v) * ldc + n;
        if (BF16OUT) ((unsigned short*)Cv)[idx] = f2bf(val);
        else         ((float*)Cv)[idx]          = val;
      }
    }
  }
}

// ---------------------------------------------------------------------------
// fp32 -> bf16 flat convert
// ---------------------------------------------------------------------------
__global__ void cvt_f32_bf16(const float* __restrict__ in,
                             unsigned short* __restrict__ out, size_t n) {
  size_t i = (size_t)blockIdx.x * blockDim.x + threadIdx.x;
  const size_t stride = (size_t)gridDim.x * blockDim.x;
  for (; i < n; i += stride) out[i] = f2bf(in[i]);
}

// ---------------------------------------------------------------------------
// fp32 [B,L,Hd] -> bf16 [B,Hd,L]  (LDS 32x33 tile)
// ---------------------------------------------------------------------------
__global__ __launch_bounds__(256) void transpose_f32_bf16(
    const float* __restrict__ in, unsigned short* __restrict__ out,
    int L, int Hd)
{
  __shared__ float tile[32][33];
  const int b  = blockIdx.z;
  const int x0 = blockIdx.x * 32;   // Hd index base
  const int y0 = blockIdx.y * 32;   // L index base
  const float*   inb  = in  + (size_t)b * L * Hd;
  unsigned short* outb = out + (size_t)b * L * Hd;
  const int tx = threadIdx.x, ty = threadIdx.y;
#pragma unroll
  for (int j = ty; j < 32; j += 8)
    tile[j][tx] = inb[(size_t)(y0 + j) * Hd + x0 + tx];
  __syncthreads();
#pragma unroll
  for (int j = ty; j < 32; j += 8)
    outb[(size_t)(x0 + j) * L + y0 + tx] = f2bf(tile[tx][j]);
}

// ---------------------------------------------------------------------------
// Row softmax over E[.,512] -> bf16 P (one wave32 per row, 16 elems/lane)
// ---------------------------------------------------------------------------
__global__ __launch_bounds__(256) void softmax_rows_k(
    const float* __restrict__ E, unsigned short* __restrict__ P)
{
  const int row  = blockIdx.x * 8 + (threadIdx.x >> 5);
  const int lane = threadIdx.x & 31;
  const float* e = E + (size_t)row * 512;
  float v[16];
  float mx = -3.402823466e38f;
#pragma unroll
  for (int i = 0; i < 16; ++i) { v[i] = e[lane + 32 * i]; mx = fmaxf(mx, v[i]); }
#pragma unroll
  for (int o = 16; o >= 1; o >>= 1) mx = fmaxf(mx, __shfl_xor(mx, o, 32));
  float s = 0.0f;
#pragma unroll
  for (int i = 0; i < 16; ++i) { v[i] = __expf(v[i] - mx); s += v[i]; }
#pragma unroll
  for (int o = 16; o >= 1; o >>= 1) s += __shfl_xor(s, o, 32);
  const float inv = 1.0f / s;
  unsigned short* p = P + (size_t)row * 512;
#pragma unroll
  for (int i = 0; i < 16; ++i) p[lane + 32 * i] = f2bf(v[i] * inv);
}

// ---------------------------------------------------------------------------
// Column softmax of E[b,:,c] (over La) -> bf16 Pt[b,c,:] (one wave per col)
// ---------------------------------------------------------------------------
__global__ __launch_bounds__(256) void softmax_cols_k(
    const float* __restrict__ E, unsigned short* __restrict__ Pt)
{
  const int idx  = blockIdx.x * 8 + (threadIdx.x >> 5);  // b*512 + c
  const int lane = threadIdx.x & 31;
  const int b = idx >> 9;
  const int c = idx & 511;
  const float* e = E + (size_t)b * 512 * 512 + c;        // stride 512
  float v[16];
  float mx = -3.402823466e38f;
#pragma unroll
  for (int i = 0; i < 16; ++i) {
    v[i] = e[(size_t)(lane + 32 * i) * 512];
    mx = fmaxf(mx, v[i]);
  }
#pragma unroll
  for (int o = 16; o >= 1; o >>= 1) mx = fmaxf(mx, __shfl_xor(mx, o, 32));
  float s = 0.0f;
#pragma unroll
  for (int i = 0; i < 16; ++i) { v[i] = __expf(v[i] - mx); s += v[i]; }
#pragma unroll
  for (int o = 16; o >= 1; o >>= 1) s += __shfl_xor(s, o, 32);
  const float inv = 1.0f / s;
  unsigned short* p = Pt + (size_t)idx * 512;
#pragma unroll
  for (int i = 0; i < 16; ++i) p[lane + 32 * i] = f2bf(v[i] * inv);
}

// ---------------------------------------------------------------------------
// Orchestration
// ---------------------------------------------------------------------------
extern "C" void kernel_launch(void* const* d_in, const int* in_sizes, int n_in,
                              void* d_out, int out_size, void* d_ws, size_t ws_size,
                              hipStream_t stream) {
  (void)in_sizes; (void)n_in; (void)out_size; (void)ws_size;

  const int Bn = 64, La = 512, Lb = 512, Hd = 1024, F = 512;
  const size_t nA = (size_t)Bn * La * Hd;   // 33,554,432 (a/b element count)
  const size_t nF = (size_t)Bn * La * F;    // 16,777,216 (fa/fb/h)
  const size_t nE = (size_t)Bn * La * Lb;   // 16,777,216 (E/P/Pt)

  const float* a  = (const float*)d_in[0];
  const float* b  = (const float*)d_in[1];
  const float* W1 = (const float*)d_in[2];
  const float* b1 = (const float*)d_in[3];
  const float* W2 = (const float*)d_in[4];
  const float* b2 = (const float*)d_in[5];

  char*  w   = (char*)d_ws;
  size_t off = 0;
  auto take = [&](size_t bytes) -> char* {
    char* p = w + off;
    off += (bytes + 255) & ~(size_t)255;
    return p;
  };

  unsigned short* aBF  = (unsigned short*)take(nA * 2);            // a bf16 [B,La,Hd]
  unsigned short* bBF  = (unsigned short*)take(nA * 2);            // b bf16 [B,Lb,Hd]
  unsigned short* aT   = (unsigned short*)take(nA * 2);            // [B,Hd,La]
  unsigned short* bT   = (unsigned short*)take(nA * 2);            // [B,Hd,Lb]
  unsigned short* W1b  = (unsigned short*)take((size_t)F * Hd * 2);
  unsigned short* W2b  = (unsigned short*)take((size_t)F * F * 2);
  unsigned short* hbuf = (unsigned short*)take(nF * 2);            // reused for a,b
  unsigned short* fa   = (unsigned short*)take(nF * 2);
  unsigned short* fb   = (unsigned short*)take(nF * 2);
  float*          E    = (float*)take(nE * 4);                     // [B,La,Lb]
  unsigned short* P    = (unsigned short*)take(nE * 2);            // row softmax
  unsigned short* Pt   = (unsigned short*)take(nE * 2);            // col softmax

  // 1) fp32 -> bf16 conversions
  cvt_f32_bf16<<<2048, 256, 0, stream>>>(a, aBF, nA);
  cvt_f32_bf16<<<2048, 256, 0, stream>>>(b, bBF, nA);
  cvt_f32_bf16<<<512, 256, 0, stream>>>(W1, W1b, (size_t)F * Hd);
  cvt_f32_bf16<<<512, 256, 0, stream>>>(W2, W2b, (size_t)F * F);

  // 2) transposes for the attend GEMMs (NT form needs [N,K] operands)
  transpose_f32_bf16<<<dim3(Hd / 32, La / 32, Bn), dim3(32, 8), 0, stream>>>(a, aT, La, Hd);
  transpose_f32_bf16<<<dim3(Hd / 32, Lb / 32, Bn), dim3(32, 8), 0, stream>>>(b, bT, Lb, Hd);

  // 3) MLP(a): h = relu(a @ W1^T + b1); fa = relu(h @ W2^T + b2)
  gemm_nt<1, 1, 1><<<dim3(F / 128, (Bn * La) / 128, 1), 128, 0, stream>>>(
      aBF, 0, W1b, 0, hbuf, 0, b1, Hd, Hd, F, Hd);
  gemm_nt<1, 1, 1><<<dim3(F / 128, (Bn * La) / 128, 1), 128, 0, stream>>>(
      hbuf, 0, W2b, 0, fa, 0, b2, F, F, F, F);

  // 4) MLP(b)
  gemm_nt<1, 1, 1><<<dim3(F / 128, (Bn * Lb) / 128, 1), 128, 0, stream>>>(
      bBF, 0, W1b, 0, hbuf, 0, b1, Hd, Hd, F, Hd);
  gemm_nt<1, 1, 1><<<dim3(F / 128, (Bn * Lb) / 128, 1), 128, 0, stream>>>(
      hbuf, 0, W2b, 0, fb, 0, b2, F, F, F, F);

  // 5) E = fa @ fb^T  (batched, fp32 out)
  gemm_nt<0, 0, 0><<<dim3(Lb / 128, La / 128, Bn), 128, 0, stream>>>(
      fa, (size_t)La * F, fb, (size_t)Lb * F, E, (size_t)La * Lb,
      nullptr, F, F, Lb, F);

  // 6) softmaxes
  softmax_rows_k<<<(Bn * La) / 8, 256, 0, stream>>>(E, P);
  softmax_cols_k<<<(Bn * Lb) / 8, 256, 0, stream>>>(E, Pt);

  // 7) beta = P @ b   -> d_out[0 .. B*La*Hd)
  float* beta  = (float*)d_out;
  float* alpha = (float*)d_out + nA;
  gemm_nt<0, 0, 0><<<dim3(Hd / 128, La / 128, Bn), 128, 0, stream>>>(
      P, (size_t)La * Lb, bT, (size_t)Hd * Lb, beta, (size_t)La * Hd,
      nullptr, Lb, Lb, Hd, Lb);

  // 8) alpha = Pt @ a -> d_out[B*La*Hd ..)
  gemm_nt<0, 0, 0><<<dim3(Hd / 128, Lb / 128, Bn), 128, 0, stream>>>(
      Pt, (size_t)Lb * La, aT, (size_t)Hd * La, alpha, (size_t)Lb * Hd,
      nullptr, La, La, Hd, La);
}
